// decoderLayer_15152644621127
// MI455X (gfx1250) — matrix operand order repaired and checked by reading, not compiled, not run on hardware
//
#include <hip/hip_runtime.h>
#include <hip/hip_bf16.h>
#include <math.h>

// ---------------------------------------------------------------------------
// Types for CDNA5 WMMA (wave32): D(16x16 f32) = A(16x32 bf16) x B(32x16 bf16) + C
// ---------------------------------------------------------------------------
typedef __bf16  v16bf  __attribute__((ext_vector_type(16)));
typedef __bf16  bf16x8 __attribute__((ext_vector_type(8)));
typedef __bf16  bf16x4 __attribute__((ext_vector_type(4)));
typedef float   v8f    __attribute__((ext_vector_type(8)));

#define WMMA_BF16(a, b, c) \
    __builtin_amdgcn_wmma_f32_16x16x32_bf16(false, (a), false, (b), (short)0, (c), false, false)

static constexpr int Bn = 2, Sn = 2048, Dn = 1024, Hn = 16, DKn = 64;
static constexpr int Mn = Bn * Sn;           // 4096 rows
static constexpr float NEGBIG = -1e30f;

// ---------------------------------------------------------------------------
// gfx1250 async copy: global -> LDS (16B per lane), tracked by ASYNCcnt.
// ---------------------------------------------------------------------------
__device__ inline void async_b128(void* lds, const void* gmem) {
    uint32_t loff = (uint32_t)(uintptr_t)lds;          // LDS aperture: low 32 bits = offset
    uint64_t ga   = (uint64_t)(uintptr_t)gmem;
    asm volatile("global_load_async_to_lds_b128 %0, %1, off"
                 :: "v"(loff), "v"(ga) : "memory");
}
__device__ inline void wait_async_0() { asm volatile("s_wait_asynccnt 0x0" ::: "memory"); }
__device__ inline void wait_async_2() { asm volatile("s_wait_asynccnt 0x2" ::: "memory"); }
__device__ inline void wait_async_4() { asm volatile("s_wait_asynccnt 0x4" ::: "memory"); }

// Build a 16-element bf16 fragment from two contiguous 8-element (16B) chunks.
__device__ inline v16bf frag_from_chunks(const __bf16* p0, const __bf16* p1) {
    bf16x8 lo = *(const bf16x8*)p0;
    bf16x8 hi = *(const bf16x8*)p1;
    v16bf a;
#pragma unroll
    for (int e = 0; e < 8; ++e) { a[e] = lo[e]; a[e + 8] = hi[e]; }
    return a;
}

// ---------------------------------------------------------------------------
// Utility kernels (vectorized)
// ---------------------------------------------------------------------------
__global__ void k_f2b(const float* __restrict__ x, __bf16* __restrict__ y, int n) {
    int i = (blockIdx.x * 256 + threadIdx.x) * 8;
    if (i >= n) return;
    float4 a = *(const float4*)(x + i);
    float4 b = *(const float4*)(x + i + 4);
    bf16x8 o;
    o[0] = (__bf16)a.x; o[1] = (__bf16)a.y; o[2] = (__bf16)a.z; o[3] = (__bf16)a.w;
    o[4] = (__bf16)b.x; o[5] = (__bf16)b.y; o[6] = (__bf16)b.z; o[7] = (__bf16)b.w;
    *(bf16x8*)(y + i) = o;
}

// Wq (H, D, dk) fp32 -> transposed bf16 Bt[(h*64+k)*D + d]
__global__ void k_pack_attn_w(const float* __restrict__ w, __bf16* __restrict__ o) {
    int i = blockIdx.x * 256 + threadIdx.x;             // i = n*D + d
    if (i >= Dn * Dn) return;
    int n = i >> 10, d = i & (Dn - 1);
    int h = n >> 6, k = n & 63;
    o[i] = (__bf16)w[(h * Dn + d) * DKn + k];
}

__global__ void k_rowmask(const float* __restrict__ x, float* __restrict__ mask) {
    __shared__ int s_any;
    int row = blockIdx.x, tid = threadIdx.x;
    if (tid == 0) s_any = 0;
    __syncthreads();
    float4 v = *(const float4*)(x + (size_t)row * Dn + tid * 4);
    if (v.x != 0.f || v.y != 0.f || v.z != 0.f || v.w != 0.f) atomicOr(&s_any, 1);
    __syncthreads();
    if (tid == 0) mask[row] = s_any ? 1.0f : 0.0f;
}

__global__ void k_add_ln(const float* __restrict__ a, const float* __restrict__ b,
                         const float* __restrict__ g, const float* __restrict__ beta,
                         float* __restrict__ outf, __bf16* __restrict__ outb) {
    __shared__ float s_sum[256], s_sq[256];
    int row = blockIdx.x, tid = threadIdx.x;
    float4 ta = *(const float4*)(a + (size_t)row * Dn + tid * 4);
    float4 tb = *(const float4*)(b + (size_t)row * Dn + tid * 4);
    float t[4] = {ta.x + tb.x, ta.y + tb.y, ta.z + tb.z, ta.w + tb.w};
    float sum = t[0] + t[1] + t[2] + t[3];
    float sq  = t[0]*t[0] + t[1]*t[1] + t[2]*t[2] + t[3]*t[3];
    s_sum[tid] = sum; s_sq[tid] = sq;
    __syncthreads();
    for (int off = 128; off > 0; off >>= 1) {
        if (tid < off) { s_sum[tid] += s_sum[tid + off]; s_sq[tid] += s_sq[tid + off]; }
        __syncthreads();
    }
    float mean = s_sum[0] * (1.0f / Dn);
    float var  = s_sq[0] * (1.0f / Dn) - mean * mean;
    float rstd = rsqrtf(var + 1e-5f);
    float4 gg = *(const float4*)(g + tid * 4);
    float4 bb = *(const float4*)(beta + tid * 4);
    float o0 = (t[0] - mean) * rstd * gg.x + bb.x;
    float o1 = (t[1] - mean) * rstd * gg.y + bb.y;
    float o2 = (t[2] - mean) * rstd * gg.z + bb.z;
    float o3 = (t[3] - mean) * rstd * gg.w + bb.w;
    if (outf) { float4 o = {o0, o1, o2, o3}; *(float4*)(outf + (size_t)row * Dn + tid * 4) = o; }
    if (outb) {
        bf16x4 o; o[0] = (__bf16)o0; o[1] = (__bf16)o1; o[2] = (__bf16)o2; o[3] = (__bf16)o3;
        *(bf16x4*)(outb + (size_t)row * Dn + tid * 4) = o;
    }
}

// ---------------------------------------------------------------------------
// Tiled BF16 WMMA GEMM with async double-buffered LDS staging.
// C[M,N] = A[M,K] x Bt[N,K]^T.  Block 256 (8 waves), BM=128 BN=128 BK=32.
// ---------------------------------------------------------------------------
template <bool BIAS, bool RELU, bool OUTF, bool OUTB, bool OUTVT>
__global__ void __launch_bounds__(256)
k_gemm_bf16(const __bf16* __restrict__ A, const __bf16* __restrict__ Bt,
            int M, int N, int K,
            const float* __restrict__ bias, const float* __restrict__ rowmask,
            float* __restrict__ Cf, __bf16* __restrict__ Cb) {
    __shared__ __bf16 sA[2][128 * 32];
    __shared__ __bf16 sB[2][128 * 32];

    const int tid  = threadIdx.x;
    const int lane = tid & 31;
    const int wave = tid >> 5;
    const int wm   = wave >> 1;
    const int wn   = wave & 1;
    const int m0   = blockIdx.y * 128;
    const int n0   = blockIdx.x * 128;
    const int lrow = lane & 15;
    const int half = lane >> 4;

    v8f acc[2][4];
#pragma unroll
    for (int i = 0; i < 2; ++i)
#pragma unroll
        for (int j = 0; j < 4; ++j) acc[i][j] = (v8f)0.0f;

    auto stage = [&](int bi, int k0) {
#pragma unroll
        for (int c = 0; c < 2; ++c) {
            int ch = tid + c * 256;
            int r  = ch >> 2, kc = (ch & 3) * 8;
            async_b128(&sA[bi][r * 32 + kc], A  + (size_t)(m0 + r) * K + k0 + kc);
            async_b128(&sB[bi][r * 32 + kc], Bt + (size_t)(n0 + r) * K + k0 + kc);
        }
    };

    const int ntiles = K >> 5;
    stage(0, 0);
    for (int t = 0; t < ntiles; ++t) {
        const int bi = t & 1;
        if (t + 1 < ntiles) { stage(bi ^ 1, (t + 1) << 5); wait_async_4(); }
        else                { wait_async_0(); }
        __syncthreads();

        v16bf af[2];
#pragma unroll
        for (int mt = 0; mt < 2; ++mt) {
            const __bf16* base = &sA[bi][(wm * 32 + mt * 16 + lrow) * 32 + half * 8];
            af[mt] = frag_from_chunks(base, base + 16);
        }
#pragma unroll
        for (int nt = 0; nt < 4; ++nt) {
            const __bf16* base = &sB[bi][(wn * 64 + nt * 16 + lrow) * 32 + half * 8];
            v16bf bf = frag_from_chunks(base, base + 16);
#pragma unroll
            for (int mt = 0; mt < 2; ++mt)
                acc[mt][nt] = WMMA_BF16(af[mt], bf, acc[mt][nt]);
        }
        __syncthreads();
    }

#pragma unroll
    for (int mt = 0; mt < 2; ++mt)
#pragma unroll
        for (int nt = 0; nt < 4; ++nt) {
            int col = n0 + wn * 64 + nt * 16 + lrow;
            float bcol = BIAS ? bias[col] : 0.0f;
#pragma unroll
            for (int r = 0; r < 8; ++r) {
                int row = m0 + wm * 32 + mt * 16 + r + 8 * half;
                float v = acc[mt][nt][r] + bcol;
                if (RELU) v = fmaxf(v, 0.0f);
                v *= rowmask[row];
                if (OUTF) Cf[(size_t)row * N + col] = v;
                if (OUTB) Cb[(size_t)row * N + col] = (__bf16)v;
                if (OUTVT) {
                    int bb = row >> 11, s = row & (Sn - 1);
                    int h  = col >> 6,  kk = col & 63;
                    Cb[((size_t)((bb * Hn + h) * DKn + kk)) * Sn + s] = (__bf16)v;
                }
            }
        }
}

// ---------------------------------------------------------------------------
// Flash attention, fully transposed dataflow (softmax lives in-lane).
//   S^T = K x Q^T   : C layout => lanes = queries, rows = keys
//   O^T = V^T x P   : C layout => lanes = queries, rows = dk
// One (b, h, 128-query block) per workgroup; 8 waves x 16 queries.
// Q,K bf16 flat [b*S][D] (head offset h*64); V pre-transposed [b][h][dk][S].
// ---------------------------------------------------------------------------
__global__ void __launch_bounds__(256)
k_attention(const __bf16* __restrict__ Q, const __bf16* __restrict__ Kb,
            const __bf16* __restrict__ Vt, const float* __restrict__ qmask,
            const float* __restrict__ kmask, float* __restrict__ Z, int causal) {
    __shared__ __bf16 sK[2][32 * 64];     // [key][dk]
    __shared__ __bf16 sV[2][64 * 32];     // [dk][key]
    __shared__ __bf16 sQ[8][16 * 64];     // per-wave Q tile [query][dk]
    __shared__ __bf16 sP[8][16 * 32];     // per-wave P^T scratch [query][key]
    __shared__ float  sO[8][16 * 17];     // per-wave output transpose (padded)

    const int tid  = threadIdx.x;
    const int lane = tid & 31;
    const int wave = tid >> 5;
    const int lrow = lane & 15;           // this lane's query column
    const int half = lane >> 4;
    const int b = blockIdx.z, h = blockIdx.y;
    const int q0 = blockIdx.x * 128;
    const int qrow_lo = q0 + wave * 16;
    const int qg = qrow_lo + lrow;        // this lane's global query index
    const size_t headoff = (size_t)h * DKn;
    const __bf16* vthead = Vt + ((size_t)(b * Hn + h) * DKn) * Sn;
    const float* kmb = kmask + b * Sn;

    // stage my wave's Q tile into LDS (async, 4 x B128 per lane)
#pragma unroll
    for (int c = 0; c < 4; ++c) {
        int ch = lane + c * 32;                          // 128 chunks per wave
        int row = ch >> 3, kc = (ch & 7) * 8;
        async_b128(&sQ[wave][row * 64 + kc],
                   Q + ((size_t)(b * Sn + qrow_lo + row)) * Dn + headoff + kc);
    }

    auto stage = [&](int bi, int kt) {
        int key = tid >> 3, kc = (tid & 7) * 8;          // K: 32 keys x 8 chunks
        async_b128(&sK[bi][key * 64 + kc],
                   Kb + ((size_t)(b * Sn + kt + key)) * Dn + headoff + kc);
        int kk = tid >> 2, keyc = (tid & 3) * 8;         // Vt: 64 dk x 4 chunks
        async_b128(&sV[bi][kk * 32 + keyc], vthead + (size_t)kk * Sn + kt + keyc);
    };

    v8f   O[4];
#pragma unroll
    for (int mt = 0; mt < 4; ++mt) O[mt] = (v8f)0.0f;
    float m = NEGBIG, l = 0.0f;           // per-lane (per-query) softmax state

    const int kend = causal ? (q0 + 128) : Sn;
    const int ntiles = kend >> 5;

    stage(0, 0);
    for (int t = 0; t < ntiles; ++t) {
        const int bi = t & 1;
        const int kt = t << 5;
        if (t + 1 < ntiles) { stage(bi ^ 1, (t + 1) << 5); wait_async_2(); }
        else                { wait_async_0(); }
        __syncthreads();

        // ---- S^T block: keys x queries.  This lane: query qg, keys in rows.
        float e[2][8];
#pragma unroll
        for (int ns = 0; ns < 2; ++ns) {
            v8f c = (v8f)0.0f;
#pragma unroll
            for (int s2 = 0; s2 < 2; ++s2) {
                const __bf16* ka = &sK[bi][(ns * 16 + lrow) * 64 + s2 * 32 + half * 8];
                v16bf ak = frag_from_chunks(ka, ka + 16);                 // A = K tile
                const __bf16* qb = &sQ[wave][lrow * 64 + s2 * 32 + half * 8];
                v16bf bq = frag_from_chunks(qb, qb + 16);                 // B = Q^T
                c = WMMA_BF16(ak, bq, c);
            }
#pragma unroll
            for (int r = 0; r < 8; ++r) e[ns][r] = c[r] * 0.125f;         // 1/sqrt(64)
        }

        // key masks: keys {8*half + r} and {16 + 8*half + r}, 4 x float4 loads
        {
            float4 a0 = *(const float4*)(kmb + kt + 8 * half);
            float4 a1 = *(const float4*)(kmb + kt + 8 * half + 4);
            float4 b0 = *(const float4*)(kmb + kt + 16 + 8 * half);
            float4 b1 = *(const float4*)(kmb + kt + 16 + 8 * half + 4);
            float km0[8] = {a0.x, a0.y, a0.z, a0.w, a1.x, a1.y, a1.z, a1.w};
            float km1[8] = {b0.x, b0.y, b0.z, b0.w, b1.x, b1.y, b1.z, b1.w};
#pragma unroll
            for (int r = 0; r < 8; ++r) {
                if (km0[r] == 0.0f) e[0][r] = NEGBIG;
                if (km1[r] == 0.0f) e[1][r] = NEGBIG;
            }
        }
        if (causal && kt + 31 > qrow_lo) {               // diagonal-straddling tiles only
#pragma unroll
            for (int r = 0; r < 8; ++r) {
                int k0i = kt + r + 8 * half;
                if (k0i > qg)      e[0][r] = NEGBIG;
                if (k0i + 16 > qg) e[1][r] = NEGBIG;
            }
        }

        // ---- online softmax: in-lane over 16, one xor-16 shuffle to merge halves
        float tm = NEGBIG;
#pragma unroll
        for (int r = 0; r < 8; ++r) tm = fmaxf(tm, fmaxf(e[0][r], e[1][r]));
        tm = fmaxf(tm, __shfl_xor(tm, 16, 32));
        float newm = fmaxf(m, tm);
        float alpha = (m <= NEGBIG * 0.5f) ? 0.0f : __expf(m - newm);
        float psum = 0.0f;
        bf16x8 p0, p1;
#pragma unroll
        for (int r = 0; r < 8; ++r) {
            float v0 = (e[0][r] <= NEGBIG * 0.5f) ? 0.0f : __expf(e[0][r] - newm);
            float v1 = (e[1][r] <= NEGBIG * 0.5f) ? 0.0f : __expf(e[1][r] - newm);
            psum += v0 + v1;
            p0[r] = (__bf16)v0; p1[r] = (__bf16)v1;
        }
        psum += __shfl_xor(psum, 16, 32);
        l = l * alpha + psum;
        m = newm;
#pragma unroll
        for (int mt = 0; mt < 4; ++mt) O[mt] *= alpha;   // in-lane rescale

        // P^T -> LDS: lane's 2x8 contiguous keys, two B128 stores
        *(bf16x8*)&sP[wave][lrow * 32 + 8 * half]      = p0;
        *(bf16x8*)&sP[wave][lrow * 32 + 16 + 8 * half] = p1;
        __syncthreads();

        // ---- O^T += V^T x P  (A = V^T tile, B = P)
        const __bf16* pb = &sP[wave][lrow * 32 + half * 8];
        v16bf bp = frag_from_chunks(pb, pb + 16);
#pragma unroll
        for (int mt = 0; mt < 4; ++mt) {
            const __bf16* va = &sV[bi][(mt * 16 + lrow) * 32 + half * 8];
            v16bf av = frag_from_chunks(va, va + 16);
            O[mt] = WMMA_BF16(av, bp, O[mt]);
        }
        __syncthreads();
    }

    // finalize: scale by qmask/l (in-lane), transpose via LDS, coalesced stores
    float inv = (l > 0.0f) ? (qmask[b * Sn + qg] / l) : 0.0f;
#pragma unroll
    for (int mt = 0; mt < 4; ++mt) {
        __syncthreads();
#pragma unroll
        for (int r = 0; r < 8; ++r)
            sO[wave][(r + 8 * half) * 17 + lrow] = O[mt][r] * inv;   // [dk][query]
        __syncthreads();
#pragma unroll
        for (int rr = 0; rr < 8; ++rr) {
            int q = qrow_lo + rr + 8 * half;
            Z[((size_t)(b * Sn + q)) * Dn + h * DKn + mt * 16 + lrow] =
                sO[wave][lrow * 17 + rr + 8 * half];
        }
    }
}

// ---------------------------------------------------------------------------
// Host-side orchestration
// ---------------------------------------------------------------------------
extern "C" void kernel_launch(void* const* d_in, const int* in_sizes, int n_in,
                              void* d_out, int out_size, void* d_ws, size_t ws_size,
                              hipStream_t stream) {
    const float* x     = (const float*)d_in[0];
    const float* ctx   = (const float*)d_in[1];
    const float* Wq1   = (const float*)d_in[2];
    const float* Wk1   = (const float*)d_in[3];
    const float* Wv1   = (const float*)d_in[4];
    const float* Wq2   = (const float*)d_in[5];
    const float* Wk2   = (const float*)d_in[6];
    const float* Wv2   = (const float*)d_in[7];
    const float* ln1_g = (const float*)d_in[8];
    const float* ln1_b = (const float*)d_in[9];
    const float* ln2_g = (const float*)d_in[10];
    const float* ln2_b = (const float*)d_in[11];
    const float* ln3_g = (const float*)d_in[12];
    const float* ln3_b = (const float*)d_in[13];
    const float* fc1_w = (const float*)d_in[14];
    const float* fc1_b = (const float*)d_in[15];
    const float* fc2_w = (const float*)d_in[16];
    const float* fc2_b = (const float*)d_in[17];
    float* out = (float*)d_out;

    char* p = (char*)d_ws;
    size_t off = 0;
    auto carve = [&](size_t bytes) { void* q = p + off; off += (bytes + 255) & ~(size_t)255; return q; };
    const size_t MD_b = (size_t)Mn * Dn * sizeof(__bf16);
    const size_t MD_f = (size_t)Mn * Dn * sizeof(float);
    const size_t DD_b = (size_t)Dn * Dn * sizeof(__bf16);

    __bf16* xb   = (__bf16*)carve(MD_b);
    __bf16* cb   = (__bf16*)carve(MD_b);
    __bf16* x1b  = (__bf16*)carve(MD_b);
    __bf16* x2b  = (__bf16*)carve(MD_b);
    __bf16* qb   = (__bf16*)carve(MD_b);
    __bf16* kb   = (__bf16*)carve(MD_b);
    __bf16* vt   = (__bf16*)carve(MD_b);   // V transposed per head [b][h][dk][S]
    __bf16* hb   = (__bf16*)carve(MD_b);
    float*  z    = (float*)carve(MD_f);
    float*  x1   = (float*)carve(MD_f);
    float*  x2   = (float*)carve(MD_f);
    float*  m0   = (float*)carve(Mn * sizeof(float));
    float*  m1   = (float*)carve(Mn * sizeof(float));
    __bf16* wq1b = (__bf16*)carve(DD_b);
    __bf16* wk1b = (__bf16*)carve(DD_b);
    __bf16* wv1b = (__bf16*)carve(DD_b);
    __bf16* wq2b = (__bf16*)carve(DD_b);
    __bf16* wk2b = (__bf16*)carve(DD_b);
    __bf16* wv2b = (__bf16*)carve(DD_b);
    __bf16* f1tb = (__bf16*)carve(DD_b);
    __bf16* f2tb = (__bf16*)carve(DD_b);

    const dim3 blk(256);
    const dim3 gemm_grid(Dn / 128, Mn / 128);
    const dim3 attn_grid(Sn / 128, Hn, Bn);
    const int nMD = Mn * Dn, nDD = Dn * Dn;
    const dim3 cvt_gridMD((nMD / 8 + 255) / 256), cvt_gridDD((nDD / 8 + 255) / 256);
    const dim3 pack_grid((nDD + 255) / 256);

    hipLaunchKernelGGL(k_f2b, cvt_gridMD, blk, 0, stream, x, xb, nMD);
    hipLaunchKernelGGL(k_f2b, cvt_gridMD, blk, 0, stream, ctx, cb, nMD);
    hipLaunchKernelGGL(k_rowmask, dim3(Mn), blk, 0, stream, x, m0);
    hipLaunchKernelGGL(k_pack_attn_w, pack_grid, blk, 0, stream, Wq1, wq1b);
    hipLaunchKernelGGL(k_pack_attn_w, pack_grid, blk, 0, stream, Wk1, wk1b);
    hipLaunchKernelGGL(k_pack_attn_w, pack_grid, blk, 0, stream, Wv1, wv1b);
    hipLaunchKernelGGL(k_pack_attn_w, pack_grid, blk, 0, stream, Wq2, wq2b);
    hipLaunchKernelGGL(k_pack_attn_w, pack_grid, blk, 0, stream, Wk2, wk2b);
    hipLaunchKernelGGL(k_pack_attn_w, pack_grid, blk, 0, stream, Wv2, wv2b);
    hipLaunchKernelGGL(k_f2b, cvt_gridDD, blk, 0, stream, fc1_w, f1tb, nDD);
    hipLaunchKernelGGL(k_f2b, cvt_gridDD, blk, 0, stream, fc2_w, f2tb, nDD);

    hipLaunchKernelGGL((k_gemm_bf16<false, false, false, true, false>), gemm_grid, blk, 0, stream,
                       xb, wq1b, Mn, Dn, Dn, (const float*)nullptr, m0, (float*)nullptr, qb);
    hipLaunchKernelGGL((k_gemm_bf16<false, false, false, true, false>), gemm_grid, blk, 0, stream,
                       xb, wk1b, Mn, Dn, Dn, (const float*)nullptr, m0, (float*)nullptr, kb);
    hipLaunchKernelGGL((k_gemm_bf16<false, false, false, false, true>), gemm_grid, blk, 0, stream,
                       xb, wv1b, Mn, Dn, Dn, (const float*)nullptr, m0, (float*)nullptr, vt);
    hipLaunchKernelGGL(k_attention, attn_grid, blk, 0, stream, qb, kb, vt, m0, m0, z, 1);
    hipLaunchKernelGGL(k_add_ln, dim3(Mn), blk, 0, stream, x, z, ln1_g, ln1_b, x1, x1b);
    hipLaunchKernelGGL(k_rowmask, dim3(Mn), blk, 0, stream, x1, m1);

    hipLaunchKernelGGL((k_gemm_bf16<false, false, false, true, false>), gemm_grid, blk, 0, stream,
                       x1b, wq2b, Mn, Dn, Dn, (const float*)nullptr, m1, (float*)nullptr, qb);
    hipLaunchKernelGGL((k_gemm_bf16<false, false, false, true, false>), gemm_grid, blk, 0, stream,
                       cb, wk2b, Mn, Dn, Dn, (const float*)nullptr, m1, (float*)nullptr, kb);
    hipLaunchKernelGGL((k_gemm_bf16<false, false, false, false, true>), gemm_grid, blk, 0, stream,
                       cb, wv2b, Mn, Dn, Dn, (const float*)nullptr, m1, (float*)nullptr, vt);
    hipLaunchKernelGGL(k_attention, attn_grid, blk, 0, stream, qb, kb, vt, m1, m1, z, 0);
    hipLaunchKernelGGL(k_add_ln, dim3(Mn), blk, 0, stream, x1, z, ln2_g, ln2_b, x2, x2b);

    hipLaunchKernelGGL((k_gemm_bf16<true, true, false, true, false>), gemm_grid, blk, 0, stream,
                       x2b, f1tb, Mn, Dn, Dn, fc1_b, m0, (float*)nullptr, hb);
    hipLaunchKernelGGL((k_gemm_bf16<true, false, true, false, false>), gemm_grid, blk, 0, stream,
                       hb, f2tb, Mn, Dn, Dn, fc2_b, m0, z, (__bf16*)nullptr);
    hipLaunchKernelGGL(k_add_ln, dim3(Mn), blk, 0, stream, x2, z, ln3_g, ln3_b, out,
                       (__bf16*)nullptr);
}
